// FlashAttentionV3_56521769616184
// MI455X (gfx1250) — matrix operand order repaired and checked by reading, not compiled, not run on hardware
//
#include <hip/hip_runtime.h>

// FlashAttention-3 style causal forward for gfx1250 (CDNA5, wave32, WMMA).
// B=2 H=16 S=2048 D=128, fp32 in/out, f16 WMMA operands, f32 accumulation.
// 64-key tiles; softmax row-sum via WMMA(ones); row-max via v_permlane16.

typedef __attribute__((ext_vector_type(16))) _Float16 v16h;
typedef __attribute__((ext_vector_type(8)))  _Float16 v8h;
typedef __attribute__((ext_vector_type(4)))  _Float16 v4h;
typedef __attribute__((ext_vector_type(8)))  float    v8f;

#define B_    2
#define H_    16
#define S_    2048
#define D_    128
#define QTILE 128          // query rows per block (8 waves x 16 rows)
#define KTILE 64           // keys per LDS stage
#define KSTR  136          // LDS stride (halves) for K tile rows  (pad: banks)
#define VSTR  72           // LDS stride (halves) for V^T rows     (pad: banks)
#define PSTR  72           // LDS stride (halves) for P scratch    (pad: banks)

__device__ __forceinline__ v16h ld16h(const _Float16* p) {
  // two 16B-aligned ds_load_b128's -> one v16h operand
  v8h lo = *reinterpret_cast<const v8h*>(p);
  v8h hi = *reinterpret_cast<const v8h*>(p + 8);
  return __builtin_shufflevector(lo, hi, 0,1,2,3,4,5,6,7,8,9,10,11,12,13,14,15);
}

// max across the 16-lane row using v_permlane16_b32 (VALU pipe, no LDS traffic;
// co-executes with XDL WMMAs). Selector nibbles implement lane ^= 1,2,4,8.
__device__ __forceinline__ float rowmax16(float x) {
  int t;
  t = __builtin_amdgcn_permlane16(0, __builtin_bit_cast(int, x),
                                  0x67452301u, 0xEFCDAB89u, true, false);
  x = fmaxf(x, __builtin_bit_cast(float, t));                      // xor 1
  t = __builtin_amdgcn_permlane16(0, __builtin_bit_cast(int, x),
                                  0x54761032u, 0xDCFE98BAu, true, false);
  x = fmaxf(x, __builtin_bit_cast(float, t));                      // xor 2
  t = __builtin_amdgcn_permlane16(0, __builtin_bit_cast(int, x),
                                  0x32107654u, 0xBA98FEDCu, true, false);
  x = fmaxf(x, __builtin_bit_cast(float, t));                      // xor 4
  t = __builtin_amdgcn_permlane16(0, __builtin_bit_cast(int, x),
                                  0xFEDCBA98u, 0x76543210u, true, false);
  x = fmaxf(x, __builtin_bit_cast(float, t));                      // xor 8
  return x;
}

__global__ __launch_bounds__(256, 1) void fa3_fwd_gfx1250(
    const float* __restrict__ Q, const float* __restrict__ K,
    const float* __restrict__ V, float* __restrict__ O) {
  __shared__ __align__(32) _Float16 Ks[KTILE * KSTR];      // [key][d]   f16
  __shared__ __align__(32) _Float16 Vt[D_ * VSTR];         // [d][key]   f16 (transposed)
  __shared__ __align__(32) _Float16 Ps[8 * 16 * PSTR];     // per-wave P scratch

  const int tid  = threadIdx.x;
  const int wave = tid >> 5;
  const int lane = tid & 31;
  const int hi   = lane >> 4;    // half-wave select
  const int ln   = lane & 15;    // N / M-within-half index

  const int nqb   = S_ / QTILE;
  const int bh    = blockIdx.x / nqb;
  const int qblk  = blockIdx.x % nqb;
  const int qbase = qblk * QTILE;
  const size_t base = (size_t)bh * S_ * D_;

  const float* Qb = Q + base;
  const float* Kb = K + base;
  const float* Vb = V + base;
  float*       Ob = O + base;

  const int qrow0 = qbase + wave * 16;   // this wave's first query row
  const int qr    = qrow0 + ln;          // A-layout row owned by this lane

  // fold softmax scale (1/sqrt(D)) and log2(e) into Q so we can use exp2f
  const float qscale = 0.08838834764831845f * 1.4426950408889634f;

  // ---- Q A-operands: 4 chunks of K=32 along D (A layout, 16x32 f16) ----
  v16h qa[4];
  {
    const float* qrow = Qb + (size_t)qr * D_;
    #pragma unroll
    for (int c = 0; c < 4; ++c) {
      const int d0 = c * 32 + hi * 16;       // halves 0..7  -> K = hi*16 + j
      const int d1 = d0 + 8;                 // halves 8..15 -> K = hi*16 + 8 + j
      const float4 a0 = *reinterpret_cast<const float4*>(qrow + d0);
      const float4 a1 = *reinterpret_cast<const float4*>(qrow + d0 + 4);
      const float4 b0 = *reinterpret_cast<const float4*>(qrow + d1);
      const float4 b1 = *reinterpret_cast<const float4*>(qrow + d1 + 4);
      qa[c][0]=(_Float16)(a0.x*qscale); qa[c][1]=(_Float16)(a0.y*qscale);
      qa[c][2]=(_Float16)(a0.z*qscale); qa[c][3]=(_Float16)(a0.w*qscale);
      qa[c][4]=(_Float16)(a1.x*qscale); qa[c][5]=(_Float16)(a1.y*qscale);
      qa[c][6]=(_Float16)(a1.z*qscale); qa[c][7]=(_Float16)(a1.w*qscale);
      qa[c][8]=(_Float16)(b0.x*qscale); qa[c][9]=(_Float16)(b0.y*qscale);
      qa[c][10]=(_Float16)(b0.z*qscale); qa[c][11]=(_Float16)(b0.w*qscale);
      qa[c][12]=(_Float16)(b1.x*qscale); qa[c][13]=(_Float16)(b1.y*qscale);
      qa[c][14]=(_Float16)(b1.z*qscale); qa[c][15]=(_Float16)(b1.w*qscale);
    }
  }

  // all-ones B operand: row-sum of P via WMMA (softmax denominator)
  v16h ones;
  #pragma unroll
  for (int j = 0; j < 16; ++j) ones[j] = (_Float16)1.0f;

  // ---- accumulators (C/D layout: M = v + 8*hi, N = ln) ----
  v8f o_acc[8];
  #pragma unroll
  for (int t = 0; t < 8; ++t) o_acc[t] = (v8f){};
  v8f l_acc = (v8f){};
  float m_run[8];
  #pragma unroll
  for (int v = 0; v < 8; ++v) m_run[v] = -1e30f;

  _Float16* pw = &Ps[wave * 16 * PSTR];

  const int nkt = (qbase + QTILE) / KTILE;   // causal: keys [0, qbase+128)
  for (int kt = 0; kt < nkt; ++kt) {
    const int kbase = kt * KTILE;
    const bool pf = (kt + 1 < nkt);

    __syncthreads();   // previous tile fully consumed
    // ---- cooperative stage: K -> Ks (f16 row-major), V -> Vt (f16 transposed)
    #pragma unroll
    for (int i = 0; i < 8; ++i) {
      const int flat = i * 256 + tid;        // float4 index, 0..2047
      const int key  = flat >> 5;            // 0..63
      const int dq   = (flat & 31) << 2;     // 0..124
      const size_t g = (size_t)(kbase + key) * D_ + dq;
      const float4 k4 = *reinterpret_cast<const float4*>(Kb + g);
      v4h kh;
      kh[0]=(_Float16)k4.x; kh[1]=(_Float16)k4.y;
      kh[2]=(_Float16)k4.z; kh[3]=(_Float16)k4.w;
      *reinterpret_cast<v4h*>(&Ks[key * KSTR + dq]) = kh;
      const float4 v4 = *reinterpret_cast<const float4*>(Vb + g);
      Vt[(dq + 0) * VSTR + key] = (_Float16)v4.x;
      Vt[(dq + 1) * VSTR + key] = (_Float16)v4.y;
      Vt[(dq + 2) * VSTR + key] = (_Float16)v4.z;
      Vt[(dq + 3) * VSTR + key] = (_Float16)v4.w;
      if (pf) {   // warm next tile into cache (global_prefetch_b8)
        __builtin_prefetch(Kb + g + (size_t)KTILE * D_, 0, 0);
        __builtin_prefetch(Vb + g + (size_t)KTILE * D_, 0, 0);
      }
    }
    __syncthreads();   // tile visible to all waves

    if (kbase > qrow0 + 15) continue;  // wave-uniform: tile fully masked

    // ---- S = Q * K^T  (four 16x16 N-subtiles, accumulate over D in 4 chunks)
    v8f s0 = (v8f){}, s1 = (v8f){}, s2 = (v8f){}, s3 = (v8f){};
    #pragma unroll
    for (int c = 0; c < 4; ++c) {
      v16h kb = ld16h(&Ks[ln * KSTR + c * 32 + hi * 16]);
      s0 = __builtin_amdgcn_wmma_f32_16x16x32_f16(false, qa[c], false, kb,
                                                  (short)0, s0, false, false);
    }
    #pragma unroll
    for (int c = 0; c < 4; ++c) {
      v16h kb = ld16h(&Ks[(16 + ln) * KSTR + c * 32 + hi * 16]);
      s1 = __builtin_amdgcn_wmma_f32_16x16x32_f16(false, qa[c], false, kb,
                                                  (short)0, s1, false, false);
    }
    #pragma unroll
    for (int c = 0; c < 4; ++c) {
      v16h kb = ld16h(&Ks[(32 + ln) * KSTR + c * 32 + hi * 16]);
      s2 = __builtin_amdgcn_wmma_f32_16x16x32_f16(false, qa[c], false, kb,
                                                  (short)0, s2, false, false);
    }
    #pragma unroll
    for (int c = 0; c < 4; ++c) {
      v16h kb = ld16h(&Ks[(48 + ln) * KSTR + c * 32 + hi * 16]);
      s3 = __builtin_amdgcn_wmma_f32_16x16x32_f16(false, qa[c], false, kb,
                                                  (short)0, s3, false, false);
    }

    // ---- causal mask (only when the tile straddles the diagonal) ----
    if (kbase + KTILE - 1 > qrow0) {
      #pragma unroll
      for (int v = 0; v < 8; ++v) {
        const int row = qrow0 + v + 8 * hi;
        if (kbase + ln      > row) s0[v] = -1e30f;
        if (kbase + 16 + ln > row) s1[v] = -1e30f;
        if (kbase + 32 + ln > row) s2[v] = -1e30f;
        if (kbase + 48 + ln > row) s3[v] = -1e30f;
      }
    }

    // ---- online softmax (log2 domain; VALU row-max, WMMA row-sum) ----
    float alpha[8];
    bool  need = false;
    #pragma unroll
    for (int v = 0; v < 8; ++v) {
      float rm = fmaxf(fmaxf(s0[v], s1[v]), fmaxf(s2[v], s3[v]));
      rm = rowmax16(rm);
      const float mn = fmaxf(m_run[v], rm);
      alpha[v] = exp2f(m_run[v] - mn);
      need = need || (mn > m_run[v]);
      m_run[v] = mn;
      s0[v] = exp2f(s0[v] - mn);
      s1[v] = exp2f(s1[v] - mn);
      s2[v] = exp2f(s2[v] - mn);
      s3[v] = exp2f(s3[v] - mn);
    }
    if (__any(need)) {        // wave-uniform: skip rescale when max unchanged
      #pragma unroll
      for (int t = 0; t < 8; ++t)
        #pragma unroll
        for (int v = 0; v < 8; ++v) o_acc[t][v] *= alpha[v];
      #pragma unroll
      for (int v = 0; v < 8; ++v) l_acc[v] *= alpha[v];
    }

    // ---- C/D-layout -> A-layout for P via per-wave LDS scratch ----
    // (LDS ops from one wave are processed in order: no barrier needed)
    #pragma unroll
    for (int v = 0; v < 8; ++v) {
      const int m = v + 8 * hi;
      pw[m * PSTR + ln]      = (_Float16)s0[v];
      pw[m * PSTR + 16 + ln] = (_Float16)s1[v];
      pw[m * PSTR + 32 + ln] = (_Float16)s2[v];
      pw[m * PSTR + 48 + ln] = (_Float16)s3[v];
    }
    v8h p00 = *reinterpret_cast<const v8h*>(&pw[ln * PSTR + hi * 16]);
    v8h p01 = *reinterpret_cast<const v8h*>(&pw[ln * PSTR + 8 + hi * 16]);
    v8h p10 = *reinterpret_cast<const v8h*>(&pw[ln * PSTR + 32 + hi * 16]);
    v8h p11 = *reinterpret_cast<const v8h*>(&pw[ln * PSTR + 40 + hi * 16]);
    v16h paA = __builtin_shufflevector(p00, p01,
                                       0,1,2,3,4,5,6,7,8,9,10,11,12,13,14,15);
    v16h paB = __builtin_shufflevector(p10, p11,
                                       0,1,2,3,4,5,6,7,8,9,10,11,12,13,14,15);

    // ---- O += P * V  (8 D-subtiles; two 32-key chunks) ----
    #pragma unroll
    for (int t = 0; t < 8; ++t) {
      v16h vb0 = ld16h(&Vt[(t * 16 + ln) * VSTR + hi * 16]);
      o_acc[t] = __builtin_amdgcn_wmma_f32_16x16x32_f16(false, paA, false, vb0,
                                                        (short)0, o_acc[t],
                                                        false, false);
      v16h vb1 = ld16h(&Vt[(t * 16 + ln) * VSTR + 32 + hi * 16]);
      o_acc[t] = __builtin_amdgcn_wmma_f32_16x16x32_f16(false, paB, false, vb1,
                                                        (short)0, o_acc[t],
                                                        false, false);
    }
    // ---- l += P * 1  (row-sum by WMMA; same layout/rescale as O) ----
    l_acc = __builtin_amdgcn_wmma_f32_16x16x32_f16(false, paA, false, ones,
                                                   (short)0, l_acc, false, false);
    l_acc = __builtin_amdgcn_wmma_f32_16x16x32_f16(false, paB, false, ones,
                                                   (short)0, l_acc, false, false);
  }

  // ---- finalize: O /= l, write f32 ----
  float rinv[8];
  #pragma unroll
  for (int v = 0; v < 8; ++v) rinv[v] = 1.0f / l_acc[v];
  #pragma unroll
  for (int t = 0; t < 8; ++t) {
    #pragma unroll
    for (int v = 0; v < 8; ++v) {
      const int row = qrow0 + v + 8 * hi;
      Ob[(size_t)row * D_ + t * 16 + ln] = o_acc[t][v] * rinv[v];
    }
  }
}

extern "C" void kernel_launch(void* const* d_in, const int* in_sizes, int n_in,
                              void* d_out, int out_size, void* d_ws, size_t ws_size,
                              hipStream_t stream) {
  (void)in_sizes; (void)n_in; (void)d_ws; (void)ws_size; (void)out_size;
  const float* Q = (const float*)d_in[0];
  const float* K = (const float*)d_in[1];
  const float* V = (const float*)d_in[2];
  float* O = (float*)d_out;
  dim3 grid(B_ * H_ * (S_ / QTILE));
  fa3_fwd_gfx1250<<<grid, 256, 0, stream>>>(Q, K, V, O);
}
